// KerasMultiLIFLayerSparseCell_67628555043244
// MI455X (gfx1250) — compile-verified, matrix operand order checked by reading, not run on hardware
//
#include <hip/hip_runtime.h>
#include <hip/hip_bf16.h>

// ---------------------------------------------------------------------------
// MI455X (gfx1250) 3-layer sparse LIF cell.
//
//   syn = C @ W^T  as f16 WMMA (C = exact integer spike counts),
//   fused LIF update, then ordered spike compaction.
//
// Data movement plan (bandwidth-bound, ~37 MB/call -> ~1.6 us at 23.3 TB/s):
//  * W (32 MB f32) streamed from HBM once (L2-resident across batch tiles),
//    converted f32->f16 in-register, prefetched ahead.
//  * A (count) tile staged into LDS once per block via the Tensor Data Mover
//    (2D D# descriptor, LDS padding for bank-conflict-free fragment reads),
//    synchronized with s_wait_tensorcnt + workgroup barrier.
// ---------------------------------------------------------------------------

typedef __attribute__((ext_vector_type(16))) _Float16 v16h;
typedef __attribute__((ext_vector_type(8)))  _Float16 v8h;
typedef __attribute__((ext_vector_type(8)))  float    v8f;
typedef __attribute__((ext_vector_type(4)))  float    v4f;
typedef __attribute__((ext_vector_type(4)))  unsigned v4u;
typedef __attribute__((ext_vector_type(8)))  int      v8i_t;
typedef __attribute__((ext_vector_type(4)))  int      v4i_t;

#define NBATCH 128
#define DECAY 0.95f
#define ONE_MINUS_DECAY 0.05f
#define THRESH 1.0f

#define KC 256                      // K-chunk staged in LDS per iteration
#define AROW 264                    // padded LDS row stride in halves (528 B)

#if defined(__has_builtin)
#if __has_builtin(__builtin_amdgcn_tensor_load_to_lds)
#define HAVE_TDM 1
#endif
#endif

// ---------------------------------------------------------------------------
// Kernel 1: build per-batch spike-count rows (f16) from (ids, num).
// One block per (layer, batch). LDS u32 counters + ds atomics, then convert.
// ---------------------------------------------------------------------------
__global__ __launch_bounds__(256) void scatter_counts(
    const int* __restrict__ inp_ids, const int* __restrict__ inp_num,
    const int* __restrict__ s1_ids,  const int* __restrict__ s1_num,
    const int* __restrict__ s2_ids,  const int* __restrict__ s2_num,
    _Float16* __restrict__ C0, _Float16* __restrict__ C1, _Float16* __restrict__ C2)
{
    __shared__ unsigned cnt[2048];

    const int layer = blockIdx.x >> 7;
    const int b     = blockIdx.x & 127;

    const int* ids; const int* num; _Float16* C; int S_in, N_in;
    if (layer == 0)      { ids = inp_ids; num = inp_num; C = C0; S_in = 128; N_in = 1024; }
    else if (layer == 1) { ids = s1_ids;  num = s1_num;  C = C1; S_in = 256; N_in = 2048; }
    else                 { ids = s2_ids;  num = s2_num;  C = C2; S_in = 256; N_in = 2048; }

    const int tid = threadIdx.x;
    for (int i = tid; i < N_in; i += 256) cnt[i] = 0u;
    __syncthreads();

    const int nb = num[b];
    for (int j = tid; j < S_in; j += 256) {
        if (j < nb) {
            int col = ids[b * S_in + j];
            atomicAdd(&cnt[col], 1u);           // ds_add_u32
        }
    }
    __syncthreads();

    for (int i = tid; i < N_in; i += 256)
        C[(size_t)b * N_in + i] = (_Float16)(float)cnt[i];
}

// ---------------------------------------------------------------------------
// Kernel 2: syn = C @ W^T with v_wmma_f32_16x16x32_f16, fused LIF update.
// grid = (16 out-tiles of 128, 8 batch-tiles of 16, 3 layers), 256 thr = 8 waves.
// A tile (16 x KC counts) staged in LDS by the TDM; W streamed from global.
//
// Fragment striping per CDNA5 ISA (16-bit A 16x32 / B 32x16):
//   lane < 16 : M/N = lane,     K in {0..7} U {16..23}
//   lane >= 16: M/N = lane-16,  K in {8..15} U {24..31}
// ---------------------------------------------------------------------------
__global__ __launch_bounds__(256) void gemm_lif(
    const _Float16* __restrict__ C0, const _Float16* __restrict__ C1, const _Float16* __restrict__ C2,
    const float* __restrict__ w0, const float* __restrict__ w1, const float* __restrict__ w2,
    const float* __restrict__ v0, const float* __restrict__ v1, const float* __restrict__ v2,
    float* __restrict__ v0n, float* __restrict__ v1n, float* __restrict__ v2n)
{
    __shared__ _Float16 Atile[16 * AROW];   // 16 rows x KC halves, 16 B pad/row

    const int layer = blockIdx.z;
    const _Float16* C; const float* W; const float* Vin; float* Vout; int N_in, N_out;
    if (layer == 0)      { C = C0; W = w0; Vin = v0; Vout = v0n; N_in = 1024; N_out = 2048; }
    else if (layer == 1) { C = C1; W = w1; Vin = v1; Vout = v1n; N_in = 2048; N_out = 2048; }
    else                 { C = C2; W = w2; Vin = v2; Vout = v2n; N_in = 2048; N_out = 1024; }

    const int tid  = threadIdx.x;
    const int wave = tid >> 5;            // wave32
    const int lane = tid & 31;
    const int half = lane >> 4;
    const int idx  = lane & 15;

    const int b0 = blockIdx.y * 16;
    const int n0 = blockIdx.x * 128 + wave * 16;
    const bool active = (n0 < N_out);     // uniform per wave

    const int koff = half ? 8 : 0;        // first K chunk for this lane-half

    const float* Wrow = W + (size_t)(n0 + idx) * N_in;   // B col n = idx -> W row
    const _Float16* Alds = Atile + idx * AROW;           // A row (M = idx) in LDS

    v8f acc = {};

    for (int kc = 0; kc < N_in; kc += KC) {
        // ---- stage A tile (16 x KC f16) into LDS -------------------------
#ifdef HAVE_TDM
        if (wave == 0) {
            const _Float16* gtile = C + (size_t)b0 * N_in + kc;
            unsigned long long ga = (unsigned long long)(const void*)gtile;
            unsigned lds_addr = (unsigned)(size_t)(const void*)&Atile[0];

            // D# group 0: count=1 | lds_addr | global_addr[56:0] | type=2
            v4u g0;
            g0.x = 1u;
            g0.y = lds_addr;
            g0.z = (unsigned)(ga & 0xFFFFFFFFull);
            g0.w = (unsigned)((ga >> 32) & 0x01FFFFFFull) | (2u << 30);

            // D# group 1: data_size=1(2B), pad_enable, pad_interval=128 DW(6),
            // pad_amount=4 DW(3); tensor_dim0=N_in, tensor_dim1=16;
            // tile_dim0=KC, tile_dim1=16; tensor_dim0_stride=N_in.
            v8i_t g1;
            g1[0] = (int)((1u << 16) | (1u << 20) | (6u << 22) | (3u << 25));
            g1[1] = (int)(((unsigned)N_in & 0xFFFFu) << 16);
            g1[2] = (int)((((unsigned)N_in >> 16) & 0xFFFFu) | (16u << 16));
            g1[3] = (int)((unsigned)KC << 16);
            g1[4] = 16;
            g1[5] = N_in;
            g1[6] = 0;
            g1[7] = 0;

            v4i_t gz4 = {0, 0, 0, 0};
#if __clang_major__ >= 23
            v8i_t gz8 = {0, 0, 0, 0, 0, 0, 0, 0};
            __builtin_amdgcn_tensor_load_to_lds(g0, g1, gz4, gz4, gz8, 0);
#else
            __builtin_amdgcn_tensor_load_to_lds(g0, g1, gz4, gz4, 0);
#endif
#if __has_builtin(__builtin_amdgcn_s_wait_tensorcnt)
            __builtin_amdgcn_s_wait_tensorcnt(0);
#else
            asm volatile("s_wait_tensorcnt 0x0" ::: "memory");
#endif
        }
#else
        // Fallback: cooperative copy, same padded layout.
        for (int t = tid; t < 16 * (KC / 8); t += 256) {
            int row = t >> 5;               // KC/8 = 32 segments of 8 halves
            int seg = t & 31;
            *(v8h*)&Atile[row * AROW + seg * 8] =
                *(const v8h*)(C + (size_t)(b0 + row) * N_in + kc + seg * 8);
        }
#endif
        __syncthreads();                    // publish A tile to all 8 waves

        if (active) {
            for (int ks = 0; ks < KC; ks += 32) {
                const int k0 = kc + ks;

                // ---- A fragment from LDS (ds_load_b128, conflict-free) ----
                v8h alo = *(const v8h*)(Alds + ks + koff);
                v8h ahi = *(const v8h*)(Alds + ks + koff + 16);
                v16h A;
#pragma unroll
                for (int i = 0; i < 8; ++i) { A[i] = alo[i]; A[8 + i] = ahi[i]; }

                // ---- B fragment: f32 weights, converted in-register ----
                v4f wa = *(const v4f*)(Wrow + k0 + koff);
                v4f wb = *(const v4f*)(Wrow + k0 + koff + 4);
                v4f wc = *(const v4f*)(Wrow + k0 + koff + 16);
                v4f wd = *(const v4f*)(Wrow + k0 + koff + 20);
                v16h B;
#pragma unroll
                for (int i = 0; i < 4; ++i) {
                    B[i]      = (_Float16)wa[i];
                    B[4 + i]  = (_Float16)wb[i];
                    B[8 + i]  = (_Float16)wc[i];
                    B[12 + i] = (_Float16)wd[i];
                }

                // Stream-ahead hint for the weight rows (global_prefetch_b8).
                __builtin_prefetch(Wrow + k0 + 256, 0, 1);

                acc = __builtin_amdgcn_wmma_f32_16x16x32_f16(
                    /*neg_a=*/false, A, /*neg_b=*/false, B,
                    /*c_mod=*/(short)0, acc, /*reuse_a=*/false, /*reuse_b=*/false);
            }
        }
        __syncthreads();                    // tile consumed; safe to overwrite
    }

    if (!active) return;

    // ---- fused LIF epilogue: reset fired, decay, add synaptic drive ----
    // D layout: VGPR r -> M = r (lanes 0-15) / M = 8+r (lanes 16-31), N = idx.
#pragma unroll
    for (int r = 0; r < 8; ++r) {
        const int bb = b0 + r + 8 * half;
        const int nn = n0 + idx;
        const size_t off = (size_t)bb * N_out + nn;
        float vo = Vin[off];
        float vv = (vo - THRESH >= 0.0f) ? 0.0f : vo;   // reset fired neurons
        Vout[off] = vv * DECAY + ONE_MINUS_DECAY * acc[r];
    }
}

// ---------------------------------------------------------------------------
// Kernel 3: ordered spike compaction (matches jax.lax.top_k on a 0/1 mask:
// fired indices ascending first, then non-fired ascending as padding).
// One block per (layer, batch); chunked inclusive scan in LDS.
// ---------------------------------------------------------------------------
__global__ __launch_bounds__(256) void compact_spikes(
    const float* __restrict__ v0n, const float* __restrict__ v1n, const float* __restrict__ v2n,
    int* __restrict__ o0_ids, int* __restrict__ o0_num,
    int* __restrict__ o1_ids, int* __restrict__ o1_num,
    int* __restrict__ o2_ids, int* __restrict__ o2_num)
{
    const int layer = blockIdx.x >> 7;
    const int b     = blockIdx.x & 127;

    const float* V; int N, S; int* oid; int* onum;
    if (layer == 0)      { V = v0n; N = 2048; S = 256; oid = o0_ids; onum = o0_num; }
    else if (layer == 1) { V = v1n; N = 2048; S = 256; oid = o1_ids; onum = o1_num; }
    else                 { V = v2n; N = 1024; S = 128; oid = o2_ids; onum = o2_num; }

    V   += (size_t)b * N;
    oid += (size_t)b * S;

    const int tid = threadIdx.x;

    __shared__ int s_tot;
    __shared__ int s_scan[256];

    if (tid == 0) s_tot = 0;
    __syncthreads();

    // Pass 1: total fired count.
    int local = 0;
    for (int i = tid; i < N; i += 256) local += (V[i] > THRESH) ? 1 : 0;
    atomicAdd(&s_tot, local);                      // LDS atomic
    __syncthreads();
    const int totf = s_tot;
    if (tid == 0) onum[b] = (totf < S) ? totf : S;

    // Pass 2: ordered placement via chunked inclusive scans (N % 256 == 0).
    int cum_f = 0;                                 // fired seen in earlier chunks
    for (int base = 0; base < N; base += 256) {
        const int i = base + tid;
        const int pred = (V[i] > THRESH) ? 1 : 0;

        s_scan[tid] = pred;
        __syncthreads();
#pragma unroll
        for (int d = 1; d < 256; d <<= 1) {
            int v   = s_scan[tid];
            int add = (tid >= d) ? s_scan[tid - d] : 0;
            __syncthreads();
            s_scan[tid] = v + add;
            __syncthreads();
        }
        const int incl   = s_scan[tid];
        const int chunkf = s_scan[255];
        const int excl_f = incl - pred;

        if (pred) {
            const int pos = cum_f + excl_f;                    // fired: ascending
            if (pos < S) oid[pos] = i;
        } else {
            const int pos = totf + i - (cum_f + excl_f);       // padding: ascending non-fired
            if (pos < S) oid[pos] = i;
        }
        cum_f += chunkf;
        __syncthreads();
    }
}

// ---------------------------------------------------------------------------
// Host launch. Inputs (setup_inputs order):
//  0 inp_ids 1 inp_num 2 s1_ids 3 s1_num 4 s2_ids 5 s2_num
//  6 v0 7 v1 8 v2 9 w0 10 w1 11 w2
// Outputs flat: o0_ids, o0_num, o1_ids, o1_num, o2_ids, o2_num, v0n, v1n, v2n
// ---------------------------------------------------------------------------
extern "C" void kernel_launch(void* const* d_in, const int* in_sizes, int n_in,
                              void* d_out, int out_size, void* d_ws, size_t ws_size,
                              hipStream_t stream)
{
    (void)in_sizes; (void)n_in; (void)out_size; (void)ws_size;

    const int*   inp_ids = (const int*)d_in[0];
    const int*   inp_num = (const int*)d_in[1];
    const int*   s1_ids  = (const int*)d_in[2];
    const int*   s1_num  = (const int*)d_in[3];
    const int*   s2_ids  = (const int*)d_in[4];
    const int*   s2_num  = (const int*)d_in[5];
    const float* v0      = (const float*)d_in[6];
    const float* v1      = (const float*)d_in[7];
    const float* v2      = (const float*)d_in[8];
    const float* w0      = (const float*)d_in[9];
    const float* w1      = (const float*)d_in[10];
    const float* w2      = (const float*)d_in[11];

    // Workspace: f16 count matrices (1.25 MB total).
    _Float16* C0 = (_Float16*)d_ws;
    _Float16* C1 = C0 + (size_t)NBATCH * 1024;
    _Float16* C2 = C1 + (size_t)NBATCH * 2048;

    // Output layout (all 4-byte elements).
    int*   o0_ids = (int*)d_out;                       // 128*256
    int*   o0_num = o0_ids + NBATCH * 256;             // 128
    int*   o1_ids = o0_num + NBATCH;                   // 128*256
    int*   o1_num = o1_ids + NBATCH * 256;             // 128
    int*   o2_ids = o1_num + NBATCH;                   // 128*128
    int*   o2_num = o2_ids + NBATCH * 128;             // 128
    float* v0n    = (float*)(o2_num + NBATCH);         // 128*2048
    float* v1n    = v0n + (size_t)NBATCH * 2048;       // 128*2048
    float* v2n    = v1n + (size_t)NBATCH * 2048;       // 128*1024

    scatter_counts<<<3 * NBATCH, 256, 0, stream>>>(
        inp_ids, inp_num, s1_ids, s1_num, s2_ids, s2_num, C0, C1, C2);

    dim3 g(16, NBATCH / 16, 3);
    gemm_lif<<<g, 256, 0, stream>>>(
        C0, C1, C2, w0, w1, w2, v0, v1, v2, v0n, v1n, v2n);

    compact_spikes<<<3 * NBATCH, 256, 0, stream>>>(
        v0n, v1n, v2n, o0_ids, o0_num, o1_ids, o1_num, o2_ids, o2_num);
}
